// GraphConvModule_39642548142690
// MI455X (gfx1250) — matrix operand order repaired and checked by reading, not compiled, not run on hardware
//
#include <hip/hip_runtime.h>
#include <hip/hip_fp16.h>

// ---------------------------------------------------------------------------
// GraphConv with filter-generating network, fused for MI455X (gfx1250).
//
// w = relu(ef@W1+b1) @ W2 done with v_wmma_f32_16x16x32_f16 (f32 accum).
// The (E,1024) intermediate never touches memory: each block computes a
// 16-edge tile's W_e in WMMA accumulators and contracts lane-locally with
// the gathered x rows (exploiting the C-matrix VGPR layout), then atomically
// aggregates into segment sums.
//
// Data movement: W2 is pre-cast to f16 AND pre-swizzled into B-fragment
// VGPR order so each fragment is a single aligned 32B load (2x b128).
// The h tile is written to LDS in A-fragment order -> 2x ds_load_b128.
// ---------------------------------------------------------------------------

typedef __attribute__((ext_vector_type(16))) _Float16 v16h;
typedef __attribute__((ext_vector_type(8)))  float    v8f;

#define N_NODES 10000
#define N_EDGES 160000
#define CIN     32
#define COUT    32
#define FDIM    13
#define HDIM    64
#define WCOLS   (CIN * COUT)   // 1024
#define TILE_E  16             // edges per block
// B fragments: 64 n-tiles x 2 k-chunks, each 32 lanes x 16 halves
#define NFRAG   (64 * 2)
#define FRAG_HALVES 512        // 32 lanes * 16 halves

// --- prep: zero sums/counts; cast+swizzle W2 into B-fragment layout ---------
// w2s[((ntile*2+kt)*32 + lane)*16 + i] = f16(W2[kt*32 + (lane>>4)*8 + (i&7) + ((i>>3)<<4),
//                                             ntile*16 + (lane&15)])
__global__ void gcn_prep_kernel(const float* __restrict__ W2,
                                float* __restrict__ sums,
                                float* __restrict__ counts,
                                _Float16* __restrict__ w2s) {
    int idx = blockIdx.x * blockDim.x + threadIdx.x;
    if (idx < N_NODES * COUT) sums[idx] = 0.0f;
    if (idx < N_NODES)        counts[idx] = 0.0f;
    if (idx < NFRAG * FRAG_HALVES) {            // 65536 halves
        int i     = idx & 15;
        int lane  = (idx >> 4) & 31;
        int frag  = idx >> 9;
        int kt    = frag & 1;
        int ntile = frag >> 1;
        int row   = kt * 32 + (lane >> 4) * 8 + (i & 7) + ((i >> 3) << 4);
        int col   = ntile * 16 + (lane & 15);
        w2s[idx] = (_Float16)W2[row * WCOLS + col];
    }
}

// --- main fused kernel: 1 block = 16 edges, 256 threads = 8 waves -----------
__launch_bounds__(256, 1)
__global__ void gcn_edge_tile_kernel(const float* __restrict__ x,
                                     const float* __restrict__ ef,
                                     const int*   __restrict__ idxn,
                                     const int*   __restrict__ dest,
                                     const float* __restrict__ W1,
                                     const float* __restrict__ b1g,
                                     const _Float16* __restrict__ w2s,
                                     const float* __restrict__ b2,
                                     float* __restrict__ sums,
                                     float* __restrict__ counts) {
    __shared__ float    sW1[FDIM][HDIM];             // 3.3 KB
    __shared__ float    sB1[HDIM];
    __shared__ float    sEf[TILE_E][FDIM + 3];       // padded
    __shared__ __align__(32) _Float16 sA[2][32][16]; // h in A-fragment order
    __shared__ float    sX[TILE_E][CIN];             // gathered x rows
    __shared__ int      sIdx[TILE_E];
    __shared__ int      sDest[TILE_E];
    __shared__ float    sPart[8][TILE_E][COUT];      // 16 KB per-wave partials

    const int tid  = threadIdx.x;
    const int base = blockIdx.x * TILE_E;

    // ---- cooperative loads --------------------------------------------------
    if (tid < TILE_E) {
        sIdx[tid]  = idxn[base + tid];
        sDest[tid] = dest[base + tid];
    }
    for (int q = tid; q < FDIM * HDIM; q += 256)
        sW1[q / HDIM][q % HDIM] = W1[q];
    if (tid < HDIM) sB1[tid] = b1g[tid];
    for (int q = tid; q < TILE_E * FDIM; q += 256)
        sEf[q / FDIM][q % FDIM] = ef[(base + q / FDIM) * FDIM + (q % FDIM)];
    __syncthreads();

    // gather source-node rows (uses sIdx)
    for (int q = tid; q < TILE_E * CIN; q += 256) {
        int m = q >> 5, i = q & 31;
        sX[m][i] = x[sIdx[m] * CIN + i];
    }
    // h = relu(ef@W1+b1) -> f16, written directly in A-fragment swizzle:
    // (e, k): kt=k>>5, kk=k&31 -> lane = e + 16*((kk>>3)&1),
    //                             idx  = (kk&7) + ((kk>>4)<<3)
    for (int q = tid; q < TILE_E * HDIM; q += 256) {
        int e = q >> 6, k = q & 63;
        float acc = sB1[k];
#pragma unroll
        for (int f = 0; f < FDIM; ++f)
            acc += sEf[e][f] * sW1[f][k];
        int kt  = k >> 5;
        int kk  = k & 31;
        int ln  = e + 16 * ((kk >> 3) & 1);
        int ix  = (kk & 7) + ((kk >> 4) << 3);
        sA[kt][ln][ix] = (_Float16)fmaxf(acc, 0.0f);
    }
    __syncthreads();

    // ---- stage 1: W_e = h @ W2 via WMMA ------------------------------------
    const int wave = tid >> 5;
    const int lane = tid & 31;
    const int c    = lane & 15;
    const int half = lane >> 4;

    // A fragments: one aligned 32B LDS load each (2x ds_load_b128)
    const v16h a0 = *(const v16h*)(&sA[0][lane][0]);
    const v16h a1 = *(const v16h*)(&sA[1][lane][0]);

    // B fragments: pre-swizzled in global, one aligned 32B load each
    const v16h* __restrict__ w2v = (const v16h*)w2s;   // 32 v16h per fragment

    v8f acc[8];
#pragma unroll
    for (int t = 0; t < 8; ++t) {
        const int ntile = wave * 8 + t;
        const v16h bf0 = w2v[(ntile * 2 + 0) * 32 + lane];
        const v16h bf1 = w2v[(ntile * 2 + 1) * 32 + lane];
        v8f cacc = {};
        cacc = __builtin_amdgcn_wmma_f32_16x16x32_f16(
            false, a0, false, bf0, (short)0, cacc, false, false);
        cacc = __builtin_amdgcn_wmma_f32_16x16x32_f16(
            false, a1, false, bf1, (short)0, cacc, false, false);
        acc[t] = cacc;
    }

    // ---- stage 2: lane-local contraction with x (exploits C layout) --------
    // C element (VGPR r, lane) = W_e[m = r + 8*half, n = wave*128 + t*16 + c]
    // n -> input channel i = wave*4 + t/2 ; output o = (t&1)*16 + c
    float bv[8];
#pragma unroll
    for (int t = 0; t < 8; ++t)
        bv[t] = b2[wave * 128 + t * 16 + c];

#pragma unroll
    for (int r = 0; r < 8; ++r) {
        const int e = r + 8 * half;
        float p0 = 0.0f, p1 = 0.0f;
#pragma unroll
        for (int j = 0; j < 4; ++j) {
            float xv = sX[e][wave * 4 + j];
            p0 += xv * (acc[2 * j][r]     + bv[2 * j]);
            p1 += xv * (acc[2 * j + 1][r] + bv[2 * j + 1]);
        }
        sPart[wave][e][c]      = p0;   // o = c      (even tiles)
        sPart[wave][e][16 + c] = p1;   // o = c + 16 (odd tiles)
    }
    __syncthreads();

    // ---- reduce 8 wave-partials, atomically aggregate into dest segments ---
#pragma unroll
    for (int qq = 0; qq < 2; ++qq) {
        int q = tid + qq * 256;        // 512 outputs: 16 edges x 32 channels
        int e = q >> 5, o = q & 31;
        float s = 0.0f;
#pragma unroll
        for (int w = 0; w < 8; ++w)
            s += sPart[w][e][o];
        atomicAdd(&sums[sDest[e] * COUT + o], s);
    }
    if (tid < TILE_E)
        atomicAdd(&counts[sDest[tid]], 1.0f);
}

// --- finalize: mean with deg==0 -> 0 ----------------------------------------
__global__ void gcn_finalize_kernel(const float* __restrict__ sums,
                                    const float* __restrict__ counts,
                                    float* __restrict__ out) {
    int idx = blockIdx.x * blockDim.x + threadIdx.x;
    if (idx < N_NODES * COUT) {
        float cnt = counts[idx >> 5];
        out[idx] = (cnt > 0.0f) ? sums[idx] / cnt : 0.0f;
    }
}

extern "C" void kernel_launch(void* const* d_in, const int* in_sizes, int n_in,
                              void* d_out, int out_size, void* d_ws, size_t ws_size,
                              hipStream_t stream) {
    const float* x    = (const float*)d_in[0];
    const float* ef   = (const float*)d_in[1];
    const int*   idxn = (const int*)d_in[2];
    const int*   dest = (const int*)d_in[3];
    const float* W1   = (const float*)d_in[4];
    const float* b1   = (const float*)d_in[5];
    const float* W2   = (const float*)d_in[6];
    const float* b2   = (const float*)d_in[7];
    float* out = (float*)d_out;

    // workspace layout (w2s offset 1,320,000 B -> 32B aligned)
    float*    sums   = (float*)d_ws;                   // N*COUT f32
    float*    counts = sums + N_NODES * COUT;          // N f32
    _Float16* w2s    = (_Float16*)(counts + N_NODES);  // 128 KB swizzled W2

    {
        int total = N_NODES * COUT;                    // covers all prep ranges
        gcn_prep_kernel<<<(total + 255) / 256, 256, 0, stream>>>(W2, sums, counts, w2s);
    }
    gcn_edge_tile_kernel<<<N_EDGES / TILE_E, 256, 0, stream>>>(
        x, ef, idxn, dest, W1, b1, w2s, b2, sums, counts);
    gcn_finalize_kernel<<<(N_NODES * COUT + 255) / 256, 256, 0, stream>>>(
        sums, counts, out);
}
